// TransformerTransducerJoiner_25520695673368
// MI455X (gfx1250) — matrix operand order repaired and checked by reading, not compiled, not run on hardware
//
#include <hip/hip_runtime.h>
#include <stdint.h>

// Problem constants (from reference)
#define B_ 8
#define T_ 256
#define U_ 128
#define H_ 512
#define I_ 640
#define V_ 1024

typedef __attribute__((ext_vector_type(16))) __bf16 v16bf;
typedef __attribute__((ext_vector_type(8)))  float  v8f;

union FragU {
    uint32_t u[8];
    uint2    u2[4];
    uint4    u4[2];
    v16bf    v;
};

__device__ __forceinline__ uint16_t f2bf(float f) {
    union { float f; uint32_t u; } x;
    x.f = f;
    uint32_t u = x.u;
    return (uint16_t)((u + 0x7FFFu + ((u >> 16) & 1u)) >> 16);  // RNE
}
__device__ __forceinline__ uint32_t pack2(float a, float b) {
    return (uint32_t)f2bf(a) | ((uint32_t)f2bf(b) << 16);
}

__device__ __forceinline__ float fast_tanh(float x) {
#if __has_builtin(__builtin_amdgcn_tanhf)
    return __builtin_amdgcn_tanhf(x);   // gfx1250 V_TANH_F32 (TRANS)
#else
    return tanhf(x);
#endif
}

// Async 16-byte global -> LDS copy (gfx1250 GLOBAL_LOAD_ASYNC_TO_LDS_B128).
// lds_off = LDS byte address (low 32 bits of the generic shared pointer,
// per the LDS aperture rule).
__device__ __forceinline__ void async_copy_b128(uint32_t lds_off, const void* gaddr) {
    asm volatile("global_load_async_to_lds_b128 %0, %1, off"
                 :
                 : "v"(lds_off), "v"((uint64_t)(uintptr_t)gaddr)
                 : "memory");
}
__device__ __forceinline__ void wait_async0() {
#if __has_builtin(__builtin_amdgcn_s_wait_asynccnt)
    __builtin_amdgcn_s_wait_asynccnt(0);
#else
    asm volatile("s_wait_asynccnt 0x0" ::: "memory");
#endif
}

// Stage a 128(n) x 32(k) bf16 tile from a transposed weight matrix
// Wt[n][k] (row stride `ldw` elements) into LDS buf [128][40] via async DMA.
// 512 x 16B segments over 256 threads = 2 per thread.
__device__ __forceinline__ void stage_b_async(uint16_t* buf, const uint16_t* __restrict__ Wt,
                                              int ldw, int n0, int k0, int tid) {
#pragma unroll
    for (int j = 0; j < 2; ++j) {
        int q   = tid + 256 * j;      // 0..511
        int row = q >> 2;             // n_local 0..127
        int seg = (q & 3) << 3;       // k_local element offset: 0,8,16,24
        uint32_t lds = (uint32_t)(uintptr_t)(&buf[row * 40 + seg]);
        const uint16_t* g = Wt + (size_t)(n0 + row) * ldw + k0 + seg;
        async_copy_b128(lds, g);
    }
}

// ---------------------------------------------------------------------------
// Kernel 0: convert + transpose W[K,N] fp32 -> Wt[N,K] bf16
// ---------------------------------------------------------------------------
__global__ __launch_bounds__(256)
void convert_transpose_kernel(const float* __restrict__ W, uint16_t* __restrict__ Wt,
                              int K, int N) {
    int idx = blockIdx.x * 256 + threadIdx.x;   // over N*K, k fastest -> coalesced writes
    int n = idx / K;
    int k = idx - n * K;
    Wt[idx] = f2bf(W[(size_t)k * N + n]);
}

// ---------------------------------------------------------------------------
// Kernel 1: out[M,640] = A[M,512] @ W[512,640] + bias   (fp32 A, bf16t W)
// 128x128 tile per workgroup, 8 waves 4(row) x 2(col), K steps of 32.
// ---------------------------------------------------------------------------
__global__ __launch_bounds__(256)
void proj_gemm_kernel(const float* __restrict__ A, const uint16_t* __restrict__ Wt,
                      const float* __restrict__ bias, float* __restrict__ out) {
    constexpr int K = H_;       // 512
    constexpr int N = I_;       // 640
    __shared__ __attribute__((aligned(16))) uint16_t Alds[128 * 40];  // [row][k]
    __shared__ __attribute__((aligned(16))) uint16_t Blds[128 * 40];  // [n][k]

    const int tid  = threadIdx.x;
    const int lane = tid & 31;
    const int wid  = tid >> 5;
    const int wRow = wid >> 1;            // 0..3
    const int wCol = wid & 1;             // 0..1
    const int half = lane >> 4;
    const int l16  = lane & 15;
    const int m0   = blockIdx.y * 128;
    const int n0   = blockIdx.x * 128;

    v8f acc[2][4];
#pragma unroll
    for (int sn = 0; sn < 4; ++sn) {
        float bv = bias[n0 + wCol * 64 + sn * 16 + l16];
#pragma unroll
        for (int sm = 0; sm < 2; ++sm)
#pragma unroll
            for (int r = 0; r < 8; ++r) acc[sm][sn][r] = bv;
    }

    for (int ks = 0; ks < K / 32; ++ks) {
        __syncthreads();                  // previous iteration's reads done
        // Stage A tile: 128 rows x 32 k (float4 load, bf16 pack, b64 LDS store)
#pragma unroll
        for (int j = 0; j < 4; ++j) {
            int q   = tid + 256 * j;      // 0..1023 float4s
            int row = q >> 3;
            int kc  = (q & 7) << 2;
            const float4 av = *(const float4*)(A + (size_t)(m0 + row) * K + ks * 32 + kc);
            uint2 p;
            p.x = pack2(av.x, av.y);
            p.y = pack2(av.z, av.w);
            *(uint2*)(&Alds[row * 40 + kc]) = p;
        }
        // Stage W tile via async DMA (already bf16, transposed)
        stage_b_async(Blds, Wt, K, n0, ks * 32, tid);
        wait_async0();
        __syncthreads();

        FragU af[2];
#pragma unroll
        for (int sm = 0; sm < 2; ++sm) {
            const uint16_t* base = &Alds[(wRow * 32 + sm * 16 + l16) * 40];
            const int kb = half * 8;
            af[sm].u2[0] = *(const uint2*)(base + kb);
            af[sm].u2[1] = *(const uint2*)(base + kb + 4);
            af[sm].u2[2] = *(const uint2*)(base + 16 + kb);
            af[sm].u2[3] = *(const uint2*)(base + 16 + kb + 4);
        }
        FragU bfr[4];
#pragma unroll
        for (int sn = 0; sn < 4; ++sn) {
            const uint16_t* base = &Blds[(wCol * 64 + sn * 16 + l16) * 40];
            bfr[sn].u4[0] = *(const uint4*)(base + half * 8);
            bfr[sn].u4[1] = *(const uint4*)(base + 16 + half * 8);
        }
#pragma unroll
        for (int sm = 0; sm < 2; ++sm)
#pragma unroll
            for (int sn = 0; sn < 4; ++sn)
                acc[sm][sn] = __builtin_amdgcn_wmma_f32_16x16x32_bf16(
                    false, af[sm].v, false, bfr[sn].v, (short)0, acc[sm][sn], false, false);
    }

#pragma unroll
    for (int sm = 0; sm < 2; ++sm)
#pragma unroll
        for (int sn = 0; sn < 4; ++sn)
#pragma unroll
            for (int r = 0; r < 8; ++r) {
                int m = m0 + wRow * 32 + sm * 16 + r + half * 8;
                int n = n0 + wCol * 64 + sn * 16 + l16;
                out[(size_t)m * N + n] = acc[sm][sn][r];
            }
}

// ---------------------------------------------------------------------------
// Kernel 2: fused tanh-joint + vocab GEMM.
// One workgroup = 128 rows = (8 t) x (16 u) of one batch, ALL of V=1024.
// Joint tile (128x640 bf16) computed once (hw V_TANH_F32) and kept resident
// in LDS (166 KB of the 320 KB WGP LDS). B tiles double-buffered via
// global_load_async_to_lds_b128 so DMA overlaps WMMA.
// ---------------------------------------------------------------------------
__global__ __launch_bounds__(256)
void joint_vocab_kernel(const float* __restrict__ encp,     // (B,T,I) fp32
                        const float* __restrict__ decp,     // (B,U,I) fp32
                        const uint16_t* __restrict__ Wt,    // (V,I) bf16 transposed
                        const float* __restrict__ bv,       // (V)
                        float* __restrict__ out) {          // (B,T,U,V)
    __shared__ __attribute__((aligned(16))) uint16_t Aj[128 * 648];     // joint bf16
    __shared__ __attribute__((aligned(16))) uint16_t Blds[2][128 * 40]; // double buffer

    const int tid  = threadIdx.x;
    const int lane = tid & 31;
    const int wid  = tid >> 5;
    const int wRow = wid >> 1;            // 0..3
    const int wCol = wid & 1;             // 0..1
    const int half = lane >> 4;
    const int l16  = lane & 15;

    const int b  = blockIdx.z;
    const int t0 = blockIdx.y * 8;
    const int u0 = blockIdx.x * 16;

    // ---- Fill joint tile: tanh(enc_proj[t] + dec_proj[u]) -> bf16 in LDS ----
#pragma unroll 4
    for (int j = 0; j < 80; ++j) {
        int q   = tid + 256 * j;          // 0..20479 float4s
        int row = q / 160;
        int c4  = (q - row * 160) * 4;
        int tt  = row >> 4, uu = row & 15;
        const float4 e = *(const float4*)(encp + (size_t)(b * T_ + t0 + tt) * I_ + c4);
        const float4 d = *(const float4*)(decp + (size_t)(b * U_ + u0 + uu) * I_ + c4);
        uint2 p;
        p.x = pack2(fast_tanh(e.x + d.x), fast_tanh(e.y + d.y));
        p.y = pack2(fast_tanh(e.z + d.z), fast_tanh(e.w + d.w));
        *(uint2*)(&Aj[row * 648 + c4]) = p;
    }
    __syncthreads();

    for (int nc = 0; nc < V_ / 128; ++nc) {          // 8 N-chunks
        const int n0 = nc * 128;
        v8f acc[2][4];
#pragma unroll
        for (int sn = 0; sn < 4; ++sn) {
            float bias = bv[n0 + wCol * 64 + sn * 16 + l16];
#pragma unroll
            for (int sm = 0; sm < 2; ++sm)
#pragma unroll
                for (int r = 0; r < 8; ++r) acc[sm][sn][r] = bias;
        }

        // Prefetch first B tile of this chunk into buffer 0.
        stage_b_async(Blds[0], Wt, I_, n0, 0, tid);

        for (int ks = 0; ks < I_ / 32; ++ks) {       // 20 K-steps
            const int cur = ks & 1;
            wait_async0();                            // my async writes landed
            __syncthreads();                          // everyone's landed; prev reads done
            if (ks + 1 < I_ / 32)                     // overlap DMA of next tile with WMMA
                stage_b_async(Blds[1 - cur], Wt, I_, n0, (ks + 1) * 32, tid);

            FragU af[2];
#pragma unroll
            for (int sm = 0; sm < 2; ++sm) {
                const uint16_t* base = &Aj[(wRow * 32 + sm * 16 + l16) * 648 + ks * 32];
                const int kb = half * 8;
                af[sm].u2[0] = *(const uint2*)(base + kb);
                af[sm].u2[1] = *(const uint2*)(base + kb + 4);
                af[sm].u2[2] = *(const uint2*)(base + 16 + kb);
                af[sm].u2[3] = *(const uint2*)(base + 16 + kb + 4);
            }
            FragU bfr[4];
#pragma unroll
            for (int sn = 0; sn < 4; ++sn) {
                const uint16_t* base = &Blds[cur][(wCol * 64 + sn * 16 + l16) * 40];
                bfr[sn].u4[0] = *(const uint4*)(base + half * 8);
                bfr[sn].u4[1] = *(const uint4*)(base + 16 + half * 8);
            }
#pragma unroll
            for (int sm = 0; sm < 2; ++sm)
#pragma unroll
                for (int sn = 0; sn < 4; ++sn)
                    acc[sm][sn] = __builtin_amdgcn_wmma_f32_16x16x32_bf16(
                        false, af[sm].v, false, bfr[sn].v, (short)0, acc[sm][sn], false, false);
        }

        // Store this 128x128 chunk of logits
#pragma unroll
        for (int sm = 0; sm < 2; ++sm)
#pragma unroll
            for (int sn = 0; sn < 4; ++sn)
#pragma unroll
                for (int r = 0; r < 8; ++r) {
                    int row = wRow * 32 + sm * 16 + r + half * 8;
                    int tt = row >> 4, uu = row & 15;
                    int n = n0 + wCol * 64 + sn * 16 + l16;
                    size_t off = (((size_t)(b * T_ + t0 + tt)) * U_ + (u0 + uu)) * V_ + n;
                    out[off] = acc[sm][sn][r];
                }
    }
}

// ---------------------------------------------------------------------------
extern "C" void kernel_launch(void* const* d_in, const int* in_sizes, int n_in,
                              void* d_out, int out_size, void* d_ws, size_t ws_size,
                              hipStream_t stream) {
    const float* enc_h   = (const float*)d_in[0];  // (B,T,H)
    const float* dec_h   = (const float*)d_in[1];  // (B,U,H)
    const float* enc_w   = (const float*)d_in[2];  // (H,I)
    const float* enc_b   = (const float*)d_in[3];  // (I)
    const float* dec_w   = (const float*)d_in[4];  // (H,I)
    const float* dec_b   = (const float*)d_in[5];  // (I)
    const float* dense_w = (const float*)d_in[6];  // (I,V)
    const float* dense_b = (const float*)d_in[7];  // (V)
    float* out = (float*)d_out;

    // Workspace layout (10 MB total)
    float*    encp     = (float*)d_ws;                          // 2048 x 640 fp32
    float*    decp     = encp + (size_t)B_ * T_ * I_;           // 1024 x 640 fp32
    uint16_t* enc_wt   = (uint16_t*)(decp + (size_t)B_ * U_ * I_); // (I,H) bf16
    uint16_t* dec_wt   = enc_wt + (size_t)I_ * H_;                 // (I,H) bf16
    uint16_t* dense_wt = dec_wt + (size_t)I_ * H_;                 // (V,I) bf16

    dim3 blk(256);
    // Pre-convert + transpose weights to bf16 [n][k]
    convert_transpose_kernel<<<dim3((I_ * H_) / 256), blk, 0, stream>>>(enc_w, enc_wt, H_, I_);
    convert_transpose_kernel<<<dim3((I_ * H_) / 256), blk, 0, stream>>>(dec_w, dec_wt, H_, I_);
    convert_transpose_kernel<<<dim3((V_ * I_) / 256), blk, 0, stream>>>(dense_w, dense_wt, I_, V_);

    // enc_proj: M=2048  |  dec_proj: M=1024  (K=512, N=640)
    proj_gemm_kernel<<<dim3(I_ / 128, (B_ * T_) / 128), blk, 0, stream>>>(
        enc_h, enc_wt, enc_b, encp);
    proj_gemm_kernel<<<dim3(I_ / 128, (B_ * U_) / 128), blk, 0, stream>>>(
        dec_h, dec_wt, dec_b, decp);

    // fused tanh-joint + vocab GEMM: grid = (U/16, T/8, B) = (8, 32, 8)
    joint_vocab_kernel<<<dim3(U_ / 16, T_ / 8, B_), blk, 0, stream>>>(
        encp, decp, dense_wt, dense_b, out);
}
